// HW_Block_Parallel_26774826123458
// MI455X (gfx1250) — matrix operand order repaired and checked by reading, not compiled, not run on hardware
//
#include <hip/hip_runtime.h>
#include <math.h>

typedef __attribute__((ext_vector_type(16))) _Float16 v16h;
typedef __attribute__((ext_vector_type(8)))  _Float16 v8h;
typedef __attribute__((ext_vector_type(8)))  float    v8f;

// ---- problem constants (match reference) ----
constexpr int Bc  = 2;
constexpr int C   = 64;
constexpr int Hh  = 128;
constexpr int Ww  = 128;
constexpr int HW  = Hh * Ww;        // 16384
constexpr int NST = 16;
constexpr int R2N = 34;             // DT_RANK + 2*N_STATE
constexpr int NPAD = 48;            // R2N padded to 3 WMMA n-tiles
constexpr int BW  = 16;             // windowed batch (B * 8)
constexpr int Lc  = 2048;           // windowed h'*w' (128*16 or 16*128)
constexpr float EPSF = 1e-6f;

__device__ __forceinline__ float sigm(float x) {
  return __builtin_amdgcn_rcpf(1.f + __expf(-x));   // v_rcp_f32, no IEEE divide
}
__device__ __forceinline__ float siluf(float x) { return x * sigm(x); }

// Build a WMMA A fragment from two contiguous 16-byte f16 segments.
__device__ __forceinline__ v16h load_a_frag(const _Float16* ap) {
  v8h lo = *(const v8h*)(ap);        // K = base+0..7
  v8h hi = *(const v8h*)(ap + 16);   // K = base+16..23
  return __builtin_shufflevector(lo, hi, 0, 1, 2, 3, 4, 5, 6, 7,
                                 8, 9, 10, 11, 12, 13, 14, 15);
}

// scan-order -> windowed-spatial index for the 4 directions
__device__ __forceinline__ int scan_src(int k, int l, int hpl, int wp) {
  int lv = (k >= 2) ? (Lc - 1 - l) : l;
  if (k & 1) {
    int hp = 1 << hpl;
    int ii = lv & (hp - 1), jj = lv >> hpl;
    return ii * wp + jj;
  }
  return lv;
}

// ---- stats: per (b,c) sum and sumsq of |x2-x1| ----
__global__ void k_stats(const float* __restrict__ x1, const float* __restrict__ x2,
                        float* __restrict__ stats) {
  int bc = blockIdx.x;
  const float* p1 = x1 + (size_t)bc * HW;
  const float* p2 = x2 + (size_t)bc * HW;
  float s1 = 0.f, s2 = 0.f;
  for (int i = threadIdx.x; i < HW; i += blockDim.x) {
    float d = fabsf(p2[i] - p1[i]);
    s1 += d; s2 += d * d;
  }
  __shared__ float r1[256], r2[256];
  r1[threadIdx.x] = s1; r2[threadIdx.x] = s2;
  __syncthreads();
  for (int off = 128; off > 0; off >>= 1) {
    if (threadIdx.x < off) { r1[threadIdx.x] += r1[threadIdx.x + off];
                             r2[threadIdx.x] += r2[threadIdx.x + off]; }
    __syncthreads();
  }
  if (threadIdx.x == 0) { stats[bc * 2] = r1[0]; stats[bc * 2 + 1] = r2[0]; }
}

// ---- att = sigmoid(g @ dar_w.T + dar_b) ----
__global__ void k_att(const float* __restrict__ stats, const float* __restrict__ dw,
                      const float* __restrict__ db, float* __restrict__ att) {
  int t = threadIdx.x;
  if (t >= Bc * C) return;
  int o = t % C, b = t / C;
  float acc = db[o];
  for (int c = 0; c < C; ++c)
    acc += (stats[(b * C + c) * 2] * (1.f / HW)) * dw[o * C + c];
  att[b * C + o] = sigm(acc);
}

// ---- x_channels(f16, pos-major) = (x1+x2)*att ; x_concat(f32) = simam ----
__global__ void k_prep(const float* __restrict__ x1, const float* __restrict__ x2,
                       const float* __restrict__ stats, const float* __restrict__ att,
                       float* __restrict__ xcat, _Float16* __restrict__ xch16) {
  size_t idx = (size_t)blockIdx.x * blockDim.x + threadIdx.x;
  if (idx >= (size_t)Bc * C * HW) return;
  int bc = (int)(idx / HW);
  int pos = (int)(idx % HW);
  int b = bc / C, c = bc % C;
  float a = x1[idx], bv = x2[idx];
  float d = fabsf(bv - a);
  xch16[((size_t)b * HW + pos) * C + c] = (_Float16)((a + bv) * att[bc]);
  float S1 = stats[bc * 2], S2 = stats[bc * 2 + 1];
  float mu = S1 * (1.f / HW);
  float v = (S2 - S1 * mu) * (1.f / (HW - 1));
  float dd = (d - mu); dd = dd * dd;
  float e_inv = dd / (4.f * (v + 1e-4f)) + 0.5f;
  xcat[idx] = d * sigm(e_inv);
}

__global__ void k_clear(float* __restrict__ p, int n) {
  int i = blockIdx.x * blockDim.x + threadIdx.x;
  if (i < n) p[i] = 0.f;
}

// x_spatial(f16, pos-major) = x_concat * branch_sum   (inputs channel-major)
__global__ void k_mul16(const float* __restrict__ a, const float* __restrict__ b,
                        _Float16* __restrict__ o, int n) {
  int i = blockIdx.x * blockDim.x + threadIdx.x;
  if (i >= n) return;
  int bc = i / HW, pos = i % HW;
  int bb = bc / C, c = bc % C;
  o[((size_t)bb * HW + pos) * C + c] = (_Float16)(a[i] * b[i]);
}

// ---- depthwise 3x3 on windowed layout (SAME pad in windowed coords) ----
// Optional dual store: f32 channel-major + f16 position-major (WMMA input).
__global__ void k_dwconv(const float* __restrict__ src, const float* __restrict__ w9,
                         const float* __restrict__ bias, float* __restrict__ dst,
                         _Float16* __restrict__ dst16,
                         int hp, int wp, int mode, int from_full, int do_silu) {
  int idx = blockIdx.x * blockDim.x + threadIdx.x;
  int total = BW * C * hp * wp;
  if (idx >= total) return;
  int jj = idx % wp; int t = idx / wp;
  int ii = t % hp;   t /= hp;
  int c  = t % C;    int bw = t / C;
  const float* wc = w9 + c * 9;
  float acc = bias[c];
  int b = bw >> 3, off = bw & 7;
  for (int di = -1; di <= 1; ++di) {
    int i2 = ii + di; if (i2 < 0 || i2 >= hp) continue;
    for (int dj = -1; dj <= 1; ++dj) {
      int j2 = jj + dj; if (j2 < 0 || j2 >= wp) continue;
      float v;
      if (from_full) {
        int fh = mode ? (i2 * 8 + off) : i2;
        int fw = mode ? j2 : (j2 * 8 + off);
        v = src[((size_t)(b * C + c) * Hh + fh) * Ww + fw];
      } else {
        v = src[(((size_t)bw * C + c) * hp + i2) * wp + j2];
      }
      acc += v * wc[(di + 1) * 3 + (dj + 1)];
    }
  }
  if (do_silu) acc = siluf(acc);
  int s = ii * wp + jj;
  dst[((size_t)bw * C + c) * (size_t)(hp * wp) + s] = acc;
  if (dst16) dst16[((size_t)bw * (hp * wp) + s) * C + c] = (_Float16)acc;
}

// ---- x_dbl projection as batched WMMA GEMM (spatial order) ----
// Out[(bw,k)][r][s] = sum_d u[bw][s][d] * xproj_w[k][r][d], r padded to 48.
// X f16 position-major; padded weights staged once per block in LDS.
__global__ void k_proj_gemm(const _Float16* __restrict__ uh,
                            const float* __restrict__ xproj,
                            float* __restrict__ pout) {
  __shared__ _Float16 sW[NPAD * C];                 // 48x64 f16 = 6KB
  int y = blockIdx.y;                               // bw*4 + k
  int bw = y >> 2, k = y & 3;
  const _Float16* Xb = uh + (size_t)bw * Lc * C;    // [Lc][C] f16
  const float* Wk = xproj + (size_t)k * R2N * C;    // [34][64] f32
  float* Ob = pout + (size_t)y * NPAD * Lc;         // [48][Lc]
  for (int i = threadIdx.x; i < NPAD * C; i += 128)
    sW[i] = (i < R2N * C) ? (_Float16)Wk[i] : (_Float16)0.f;
  __syncthreads();
  int wave = threadIdx.x >> 5, lane = threadIdx.x & 31;
  int tile = blockIdx.x * 4 + wave;                 // 384 tiles = 128m x 3n
  int nt = tile % 3, mt = tile / 3;
  int m0 = mt << 4, n0 = nt << 4;
  int ml = lane & 15, half = lane >> 4;
  int nrow = n0 + ml;
  const _Float16* arow = Xb + (size_t)(m0 + ml) * C + half * 8;
  v8f acc = {};
  #pragma unroll
  for (int k0 = 0; k0 < C; k0 += 32) {
    v16h a = load_a_frag(arow + k0);                // two b128 loads
    v16h bm;
    #pragma unroll
    for (int e = 0; e < 16; ++e)
      bm[e] = sW[nrow * C + k0 + half * 8 + (e < 8 ? e : e + 8)];
    acc = __builtin_amdgcn_wmma_f32_16x16x32_f16(false, a, false, bm,
                                                 (short)0, acc, false, false);
  }
  #pragma unroll
  for (int p = 0; p < 8; ++p)
    Ob[(size_t)nrow * Lc + m0 + half * 8 + p] = acc[p];
}

// ---- dt expansion + softplus + pack {dt[64], B[16], C[16]} per (bw,k,s) ----
__global__ void k_dt(const float* __restrict__ pglob, const float* __restrict__ dtw,
                     const float* __restrict__ dtb, float* __restrict__ pack) {
  __shared__ float sDtw[C * 2];
  __shared__ float sDtb[C];
  int idx = blockIdx.x * 256 + threadIdx.x;   // over BW*4*Lc, y uniform per block
  int s = idx & (Lc - 1);
  int y = idx >> 11;
  int k = y & 3;
  for (int i = threadIdx.x; i < C * 2; i += 256) sDtw[i] = dtw[k * C * 2 + i];
  for (int i = threadIdx.x; i < C;     i += 256) sDtb[i] = dtb[k * C + i];
  __syncthreads();
  const float* pg = pglob + (size_t)y * NPAD * Lc;
  float xd0 = pg[s], xd1 = pg[Lc + s];
  float* pp = pack + ((size_t)y * Lc + s) * 96;
  for (int d = 0; d < C; ++d) {
    float z = xd0 * sDtw[d * 2] + xd1 * sDtw[d * 2 + 1] + sDtb[d];
    pp[d] = (z > 20.f) ? z : log1pf(__expf(z));
  }
  #pragma unroll
  for (int n = 0; n < NST; ++n) {
    pp[64 + n] = pg[(size_t)(2 + n) * Lc + s];
    pp[80 + n] = pg[(size_t)(18 + n) * Lc + s];
  }
}

// ---- selective scan: one block per (bw,k), one thread per channel d ----
__global__ void k_scan(const float* __restrict__ u, const float* __restrict__ pack,
                       const float* __restrict__ Alogs, float* __restrict__ ys,
                       int hpl, int wp) {
  int k = blockIdx.x & 3, bw = blockIdx.x >> 2;
  int d = threadIdx.x;
  __shared__ float sB[NST], sC[NST];
  float A[NST], hst[NST];
  #pragma unroll
  for (int n = 0; n < NST; ++n) {
    A[n] = -__expf(Alogs[(k * C + d) * NST + n]);
    hst[n] = 0.f;
  }
  const float* ub = u + (size_t)bw * C * Lc + (size_t)d * Lc;
  const float* pb = pack + (size_t)(bw * 4 + k) * Lc * 96;
  float* yb = ys + ((size_t)(bw * 4 + k) * C + d) * Lc;
  for (int l = 0; l < Lc; ++l) {
    int s = scan_src(k, l, hpl, wp);
    const float* pl = pb + (size_t)s * 96;
    if (threadIdx.x < NST)            sB[threadIdx.x]       = pl[64 + threadIdx.x];
    else if (threadIdx.x < 2 * NST)   sC[threadIdx.x - NST] = pl[80 + threadIdx.x - NST];
    __syncthreads();
    float dt = pl[d];
    float xt = ub[s];
    float dx = dt * xt;
    float y = 0.f;
    #pragma unroll
    for (int n = 0; n < NST; ++n) {
      hst[n] = hst[n] * __expf(dt * A[n]) + dx * sB[n];
      y += hst[n] * sC[n];
    }
    yb[l] = y;
    __syncthreads();
  }
}

// ---- combine 4 directions + D*x, two channel-LNs, un-window into bsum ----
__global__ void k_combine(const float* __restrict__ ys, const float* __restrict__ u,
                          const float* __restrict__ Ds, const float* __restrict__ ng,
                          const float* __restrict__ nb, const float* __restrict__ bg,
                          const float* __restrict__ bb, float* __restrict__ bsum,
                          int hpl, int wpl, int mode) {
  __shared__ float sv[C * 64];
  __shared__ float sDsum[C];
  int t = threadIdx.x; // 64 threads, one position each
  sDsum[t] = Ds[t] + Ds[C + t] + Ds[2 * C + t] + Ds[3 * C + t];
  __syncthreads();
  int idx = blockIdx.x * 64 + t;
  int s  = idx & (Lc - 1);
  int bw = idx >> 11;
  int hp = 1 << hpl, wp = 1 << wpl;
  int ii = s >> wpl, jj = s & (wp - 1);
  int l0 = s, l1 = jj * hp + ii;
  const float* ysb = ys + (size_t)bw * 4 * C * Lc;
  const float* ub  = u  + (size_t)bw * C * Lc;
  for (int d = 0; d < C; ++d) {
    const float* y0 = ysb + (size_t)d * Lc;
    const float* y1 = ysb + (size_t)(C + d) * Lc;
    const float* y2 = ysb + (size_t)(2 * C + d) * Lc;
    const float* y3 = ysb + (size_t)(3 * C + d) * Lc;
    float v = y0[l0] + y1[l1] + y2[Lc - 1 - l0] + y3[Lc - 1 - l1]
            + ub[(size_t)d * Lc + s] * sDsum[d];
    sv[d * 64 + t] = v;
  }
  // LN1 (ss2d norm)
  float mu = 0.f, s2 = 0.f;
  for (int d = 0; d < C; ++d) { float v = sv[d * 64 + t]; mu += v; s2 += v * v; }
  mu *= (1.f / C); float var = s2 * (1.f / C) - mu * mu;
  float rs = rsqrtf(var + EPSF);
  for (int d = 0; d < C; ++d)
    sv[d * 64 + t] = (sv[d * 64 + t] - mu) * rs * ng[d] + nb[d];
  // LN2 (branch norm)
  mu = 0.f; s2 = 0.f;
  for (int d = 0; d < C; ++d) { float v = sv[d * 64 + t]; mu += v; s2 += v * v; }
  mu *= (1.f / C); var = s2 * (1.f / C) - mu * mu;
  rs = rsqrtf(var + EPSF);
  int b = bw >> 3, off = bw & 7;
  int fh = mode ? (ii * 8 + off) : ii;
  int fw = mode ? jj : (jj * 8 + off);
  for (int d = 0; d < C; ++d) {
    float v = (sv[d * 64 + t] - mu) * rs * bg[d] + bb[d];
    bsum[((size_t)(b * C + d) * Hh + fh) * Ww + fw] += v;
  }
}

// ---- channel LayerNorm (optional add, optional silu) ----
// f32 output channel-major; f16 output position-major (WMMA input).
__global__ void k_ln_chan(const float* __restrict__ a, const float* __restrict__ b2,
                          const float* __restrict__ g, const float* __restrict__ bt,
                          float* __restrict__ out, _Float16* __restrict__ out16,
                          int do_silu) {
  int idx = blockIdx.x * blockDim.x + threadIdx.x;
  if (idx >= Bc * HW) return;
  int pos = idx % HW, bb = idx / HW;
  const float* ap = a + (size_t)bb * C * HW + pos;
  const float* bp = b2 ? (b2 + (size_t)bb * C * HW + pos) : nullptr;
  float mu = 0.f, s2 = 0.f;
  for (int c = 0; c < C; ++c) {
    float v = ap[(size_t)c * HW]; if (bp) v += bp[(size_t)c * HW];
    mu += v; s2 += v * v;
  }
  mu *= (1.f / C); float var = s2 * (1.f / C) - mu * mu;
  float rs = rsqrtf(var + EPSF);
  size_t ocm = (size_t)bb * C * HW + pos;              // channel-major base
  size_t opm = ((size_t)bb * HW + pos) * C;            // position-major base
  for (int c = 0; c < C; ++c) {
    float v = ap[(size_t)c * HW]; if (bp) v += bp[(size_t)c * HW];
    v = (v - mu) * rs * g[c] + bt[c];
    if (do_silu) v = siluf(v);
    if (out)   out[ocm + (size_t)c * HW] = v;
    if (out16) out16[opm + c] = (_Float16)v;
  }
}

// ---- WMMA GEMM: Out[n][m] = sum_k X[m][k] * W[n][k] + bias[n], per batch ----
// X f16 position-major (M x K): A fragment = two b128 loads per chunk.
// W f32 row-major (N x K) staged to LDS as f16.
// Of: f32 channel-major (N x M). Oh: f16 position-major (M x N).
template <int K>
__global__ void k_gemm_wmma(const _Float16* __restrict__ X, const float* __restrict__ Wt,
                            const float* __restrict__ bias, float* __restrict__ Of,
                            _Float16* __restrict__ Oh, int M, int N, int silu) {
  __shared__ _Float16 sW[256 * 64];                // max N*K = 16384 f16 = 32KB
  int b = blockIdx.y;
  const _Float16* Xb = X + (size_t)b * M * K;
  for (int i = threadIdx.x; i < N * K; i += 128) sW[i] = (_Float16)Wt[i];
  __syncthreads();
  int wave = threadIdx.x >> 5;
  int lane = threadIdx.x & 31;
  int tilesN = N >> 4;
  int tile = blockIdx.x * 4 + wave;
  int totTiles = (M >> 4) * tilesN;
  if (tile >= totTiles) return;              // wave-uniform: EXEC stays all-ones
  int nt = tile % tilesN, mt = tile / tilesN;
  int m0 = mt << 4, n0 = nt << 4;
  int ml = lane & 15, half = lane >> 4;
  const _Float16* arow = Xb + (size_t)(m0 + ml) * K + half * 8;
  v8f acc = {};
  #pragma unroll
  for (int k0 = 0; k0 < K; k0 += 32) {
    if (k0 + 32 < K) __builtin_prefetch(arow + k0 + 32, 0, 3);
    v16h a = load_a_frag(arow + k0);         // two b128 loads
    v16h bm;
    #pragma unroll
    for (int e = 0; e < 16; ++e)
      bm[e] = sW[(n0 + ml) * K + k0 + half * 8 + (e < 8 ? e : e + 8)];
    acc = __builtin_amdgcn_wmma_f32_16x16x32_f16(false, a, false, bm,
                                                 (short)0, acc, false, false);
  }
  float bs = bias[n0 + ml];
  #pragma unroll
  for (int p = 0; p < 8; ++p) {
    // D: VGPR p holds row M = p + 8*half, col N = lane%16
    float v = acc[p] + bs;
    if (silu) v = siluf(v);
    if (Of) Of[(size_t)b * N * M + (size_t)(n0 + ml) * M + m0 + half * 8 + p] = v;
    if (Oh) Oh[((size_t)b * M + m0 + half * 8 + p) * N + n0 + ml] = (_Float16)v;
  }
}

extern "C" void kernel_launch(void* const* d_in, const int* in_sizes, int n_in,
                              void* d_out, int out_size, void* d_ws, size_t ws_size,
                              hipStream_t stream) {
  (void)in_sizes; (void)n_in; (void)out_size; (void)ws_size;
  const float* x1        = (const float*)d_in[0];
  const float* x2        = (const float*)d_in[1];
  const float* br_dw_w   = (const float*)d_in[2];
  const float* br_dw_b   = (const float*)d_in[3];
  const float* ss_conv_w = (const float*)d_in[4];
  const float* ss_conv_b = (const float*)d_in[5];
  const float* ss_xproj  = (const float*)d_in[6];
  const float* ss_dt_w   = (const float*)d_in[7];
  const float* ss_dt_b   = (const float*)d_in[8];
  const float* ss_Alogs  = (const float*)d_in[9];
  const float* ss_Ds     = (const float*)d_in[10];
  const float* ss_ng     = (const float*)d_in[11];
  const float* ss_nb     = (const float*)d_in[12];
  const float* br_ln_g   = (const float*)d_in[13];
  const float* br_ln_b   = (const float*)d_in[14];
  const float* dar_w     = (const float*)d_in[15];
  const float* dar_b     = (const float*)d_in[16];
  const float* cc_w      = (const float*)d_in[17];
  const float* cc_b      = (const float*)d_in[18];
  const float* cc_ln_g   = (const float*)d_in[19];
  const float* cc_ln_b   = (const float*)d_in[20];
  const float* sc_w      = (const float*)d_in[21];
  const float* sc_b      = (const float*)d_in[22];
  const float* sc_ln_g   = (const float*)d_in[23];
  const float* sc_ln_b   = (const float*)d_in[24];
  const float* mlp_ln_g  = (const float*)d_in[25];
  const float* mlp_ln_b  = (const float*)d_in[26];
  const float* mlp_w1    = (const float*)d_in[27];
  const float* mlp_b1    = (const float*)d_in[28];
  const float* mlp_w2    = (const float*)d_in[29];
  const float* mlp_b2    = (const float*)d_in[30];
  float* out = (float*)d_out;

  const int nel = Bc * C * HW;                       // 2,097,152

  // workspace layout (float-sized slots; f16 regions cast; all 16B aligned)
  float* ws    = (float*)d_ws;
  float* stats = ws;                                 // Bc*C*2
  float* att   = stats + (size_t)Bc * C * 2;         // Bc*C
  float* xcat  = att + (size_t)Bc * C;               // nel
  float* bsum  = xcat + nel;                         // nel
  float* ybuf  = bsum + nel;                         // BW*C*L
  float* ubuf  = ybuf + (size_t)BW * C * Lc;         // BW*C*L
  float* pack  = ubuf + (size_t)BW * C * Lc;         // BW*4*L*96
  float* ysb   = pack + (size_t)BW * 4 * Lc * 96;    // BW*4*C*L
  float* pglob = ysb + (size_t)BW * 4 * C * Lc;      // BW*4*NPAD*L
  float* f16z  = pglob + (size_t)BW * 4 * NPAD * Lc; // f16 zone start
  _Float16* uh      = (_Float16*)f16z;               // BW*L*C halves (pos-major)
  _Float16* xch16   = (_Float16*)(f16z + (size_t)BW * C * Lc / 2);
  _Float16* xsp16   = xch16 + nel;
  _Float16* hfeat16 = xsp16 + nel;
  _Float16* m116    = hfeat16 + nel;                 // Bc*HW*4C halves (pos-major)
  // reuse after branch phase (f32):
  float* t1    = ybuf;
  float* t2    = ubuf;
  float* chan  = pack;
  float* spat  = pack + nel;

  k_stats<<<Bc * C, 256, 0, stream>>>(x1, x2, stats);
  k_att<<<1, 128, 0, stream>>>(stats, dar_w, dar_b, att);
  k_prep<<<nel / 256, 256, 0, stream>>>(x1, x2, stats, att, xcat, xch16);
  k_clear<<<nel / 256, 256, 0, stream>>>(bsum, nel);

  for (int i = 0; i < 4; ++i) {
    const float* src = (i == 0 || i == 2) ? x1 : x2;
    int mode = (i >= 2) ? 1 : 0;
    int hp  = mode ? 16 : 128, wp  = mode ? 128 : 16;
    int hpl = mode ? 4  : 7,   wpl = mode ? 7   : 4;
    k_dwconv<<<(BW * C * Lc) / 256, 256, 0, stream>>>(
        src, br_dw_w + i * C * 9, br_dw_b + i * C, ybuf, nullptr,
        hp, wp, mode, 1, 0);
    k_dwconv<<<(BW * C * Lc) / 256, 256, 0, stream>>>(
        ybuf, ss_conv_w + i * C * 9, ss_conv_b + i * C, ubuf, uh,
        hp, wp, mode, 0, 1);
    // x_dbl projection on WMMA (spatial order, batched over (bw,k))
    k_proj_gemm<<<dim3(96, BW * 4), 128, 0, stream>>>(
        uh, ss_xproj + (size_t)i * 4 * R2N * C, pglob);
    k_dt<<<(BW * 4 * Lc) / 256, 256, 0, stream>>>(
        pglob, ss_dt_w + (size_t)i * 4 * C * 2, ss_dt_b + (size_t)i * 4 * C, pack);
    k_scan<<<BW * 4, 64, 0, stream>>>(
        ubuf, pack, ss_Alogs + (size_t)i * 4 * C * NST, ysb, hpl, wp);
    k_combine<<<(BW * Lc) / 64, 64, 0, stream>>>(
        ysb, ubuf, ss_Ds + (size_t)i * 4 * C, ss_ng + i * C, ss_nb + i * C,
        br_ln_g + i * C, br_ln_b + i * C, bsum, hpl, wpl, mode);
  }

  k_mul16<<<nel / 256, 256, 0, stream>>>(xcat, bsum, xsp16, nel);

  dim3 gN64((HW / 16) * (C / 16) / 4, Bc);          // 1024 blocks x 4 waves
  dim3 gN256((HW / 16) * (4 * C / 16) / 4, Bc);     // 4096 blocks x 4 waves
  k_gemm_wmma<C><<<gN64, 128, 0, stream>>>(xch16, cc_w, cc_b, t1, nullptr, HW, C, 0);
  k_ln_chan<<<(Bc * HW) / 256, 256, 0, stream>>>(t1, nullptr, cc_ln_g, cc_ln_b,
                                                 chan, nullptr, 1);
  k_gemm_wmma<C><<<gN64, 128, 0, stream>>>(xsp16, sc_w, sc_b, t2, nullptr, HW, C, 0);
  k_ln_chan<<<(Bc * HW) / 256, 256, 0, stream>>>(t2, nullptr, sc_ln_g, sc_ln_b,
                                                 spat, nullptr, 1);
  k_ln_chan<<<(Bc * HW) / 256, 256, 0, stream>>>(chan, spat, mlp_ln_g, mlp_ln_b,
                                                 nullptr, hfeat16, 0);
  k_gemm_wmma<C><<<gN256, 128, 0, stream>>>(hfeat16, mlp_w1, mlp_b1,
                                            nullptr, m116, HW, 4 * C, 1);
  k_gemm_wmma<4 * C><<<gN64, 128, 0, stream>>>(m116, mlp_w2, mlp_b2,
                                               out, nullptr, HW, C, 0);
}
